// JointTransPointAbstraction_33655363732088
// MI455X (gfx1250) — matrix operand order repaired and checked by reading, not compiled, not run on hardware
//
#include <hip/hip_runtime.h>

#define DEVI __device__ __forceinline__

typedef _Float16 half_t;
typedef __attribute__((ext_vector_type(16))) _Float16 v16h;
typedef __attribute__((ext_vector_type(8)))  _Float16 v8h;
typedef __attribute__((ext_vector_type(8)))  float    v8f;

constexpr int B_ = 8, N_ = 8192, NPOINT = 2048, NSAMPLE = 32, DIN = 61, DD = 64;
constexpr float RADIUS = 0.2f;

DEVI int lane_id() { return threadIdx.x & 31; }

// ---- WMMA fragment helpers (wave32, ISA C/D layout, A/B with contiguous k = 16*(lane>>4) + s) ----
// All f16 sources are laid out so each lane's fragment share is contiguous,
// 16B-aligned -> b128 vector loads.

DEVI v16h cat8(v8h lo, v8h hi) {
  return __builtin_shufflevector(lo, hi, 0, 1, 2, 3, 4, 5, 6, 7, 8, 9, 10, 11,
                                 12, 13, 14, 15);
}

// A fragment: 16x32 f16 tile, row-major, row stride ld halves (ld*2 % 16 == 0)
DEVI v16h load_A16(const half_t* base, int ld) {
  int l = lane_id();
  const half_t* p = base + (l & 15) * ld + ((l >> 4) * 16);
  return cat8(*(const v8h*)p, *(const v8h*)(p + 8));
}

// B fragment: 32x16 (K x N), element B[k][n] = base[n*ld + k] (n-major source)
DEVI v16h load_Bc(const half_t* base, int ld) {
  int l = lane_id();
  const half_t* p = base + (l & 15) * ld + ((l >> 4) * 16);
  return cat8(*(const v8h*)p, *(const v8h*)(p + 8));
}

// C/D fragment: 16x16 f32, row = j + 8*(lane>>4), col = lane&15
DEVI v8f load_C32(const float* base, int ld) {
  int l = lane_id();
  const float* p = base + ((l >> 4) * 8) * ld + (l & 15);
  v8f c;
#pragma unroll
  for (int j = 0; j < 8; ++j) c[j] = p[j * ld];
  return c;
}
DEVI void store_C32(float* base, int ld, v8f c) {
  int l = lane_id();
  float* p = base + ((l >> 4) * 8) * ld + (l & 15);
#pragma unroll
  for (int j = 0; j < 8; ++j) p[j * ld] = c[j];
}
DEVI void store_C16(half_t* base, int ld, v8f c) {
  int l = lane_id();
  half_t* p = base + ((l >> 4) * 8) * ld + (l & 15);
#pragma unroll
  for (int j = 0; j < 8; ++j) p[j * ld] = (half_t)c[j];
}

DEVI v8f wmma16(v16h a, v16h b, v8f c) {
  return __builtin_amdgcn_wmma_f32_16x16x32_f16(false, a, false, b, (short)0, c,
                                                false, false);
}

// ======================= 1. Farthest Point Sampling =======================
__global__ __launch_bounds__(1024) void fps_kernel(
    const float* __restrict__ xyz, int* __restrict__ fidx,
    float* __restrict__ nxyz, float* __restrict__ out0) {
  const int b = blockIdx.x, t = threadIdx.x;
  const float* xb = xyz + (size_t)b * 3 * N_;
  float lx[8], ly[8], lz[8], ldst[8];
#pragma unroll
  for (int i = 0; i < 8; ++i) {
    int n = t + i * 1024;
    lx[i] = xb[n]; ly[i] = xb[N_ + n]; lz[i] = xb[2 * N_ + n];
    ldst[i] = 1e10f;
  }
  __shared__ float scx, scy, scz;
  __shared__ int sfar;
  __shared__ float wmax[32];
  __shared__ int warg[32];
  if (t == 0) { sfar = 0; scx = xb[0]; scy = xb[N_]; scz = xb[2 * N_]; }
  __syncthreads();

  for (int it = 0; it < NPOINT; ++it) {
    const int far = sfar;
    const float cx = scx, cy = scy, cz = scz;
    if (t == 0) {
      fidx[b * NPOINT + it] = far;
      nxyz[((size_t)b * NPOINT + it) * 3 + 0] = cx;
      nxyz[((size_t)b * NPOINT + it) * 3 + 1] = cy;
      nxyz[((size_t)b * NPOINT + it) * 3 + 2] = cz;
      out0[((size_t)b * 3 + 0) * NPOINT + it] = cx;
      out0[((size_t)b * 3 + 1) * NPOINT + it] = cy;
      out0[((size_t)b * 3 + 2) * NPOINT + it] = cz;
    }
    float m = -1.0f; int a = 0;
#pragma unroll
    for (int i = 0; i < 8; ++i) {
      float dx = lx[i] - cx, dy = ly[i] - cy, dz = lz[i] - cz;
      float d = dx * dx + dy * dy + dz * dz;
      ldst[i] = fminf(ldst[i], d);
      int n = t + i * 1024;
      if (ldst[i] > m || (ldst[i] == m && n < a)) { m = ldst[i]; a = n; }
    }
#pragma unroll
    for (int off = 16; off >= 1; off >>= 1) {
      float om = __shfl_xor(m, off, 32);
      int   oa = __shfl_xor(a, off, 32);
      if (om > m || (om == m && oa < a)) { m = om; a = oa; }
    }
    if (lane_id() == 0) { wmax[t >> 5] = m; warg[t >> 5] = a; }
    __syncthreads();
    if (t < 32) {
      m = wmax[t]; a = warg[t];
#pragma unroll
      for (int off = 16; off >= 1; off >>= 1) {
        float om = __shfl_xor(m, off, 32);
        int   oa = __shfl_xor(a, off, 32);
        if (om > m || (om == m && oa < a)) { m = om; a = oa; }
      }
      if (t == 0) sfar = a;
    }
    __syncthreads();
    const int nf = sfar;
    if ((nf & 1023) == t) {
#pragma unroll
      for (int i = 0; i < 8; ++i)
        if ((nf >> 10) == i) { scx = lx[i]; scy = ly[i]; scz = lz[i]; }
    }
    __syncthreads();
  }
}

// ======================= 2. Ball query =======================
__global__ __launch_bounds__(256) void ballq_kernel(
    const float* __restrict__ xyz, const float* __restrict__ nxyz,
    int* __restrict__ gi) {
  const int w = threadIdx.x >> 5, l = threadIdx.x & 31;
  const int idx = blockIdx.x * 8 + w;
  const int b = idx / NPOINT;
  const float* xb = xyz + (size_t)b * 3 * N_;
  const float cx = nxyz[(size_t)idx * 3 + 0];
  const float cy = nxyz[(size_t)idx * 3 + 1];
  const float cz = nxyz[(size_t)idx * 3 + 2];
  int* grow = gi + (size_t)idx * NSAMPLE;
  int cnt = 0, first = 0;
  for (int base = 0; base < N_; base += 32) {
    int n = base + l;
    float dx = xb[n] - cx, dy = xb[N_ + n] - cy, dz = xb[2 * N_ + n] - cz;
    bool pred = (dx * dx + dy * dy + dz * dz) <= (RADIUS * RADIUS);
    unsigned mask = (unsigned)__ballot(pred);
    if (cnt == 0 && mask) first = base + __builtin_ctz(mask);
    int pos = __builtin_popcount(mask & ((1u << l) - 1u));
    if (pred && (cnt + pos) < NSAMPLE) grow[cnt + pos] = n;
    cnt += __builtin_popcount(mask);
    if (cnt >= NSAMPLE) break;
  }
  if (l >= cnt) grow[l] = first;  // pad (cnt>=1: center is itself a point)
}

// ======================= 3. Weight f16 prep (transposed: wT[n][k]) ========
__global__ void wprep_kernel(const float* s0, const float* s1, const float* s2,
                             const float* s3, const float* s4, const float* s5,
                             const float* s6, const float* s7,
                             half_t* __restrict__ dst) {
  int e = blockIdx.x * blockDim.x + threadIdx.x;
  if (e >= 8 * 4096) return;
  const float* srcs[8] = {s0, s1, s2, s3, s4, s5, s6, s7};
  int m = e >> 12, r = e & 4095;
  int n = r >> 6, k = r & 63;
  dst[e] = (half_t)srcs[m][k * 64 + n];  // dst[m][n*64+k] = w[m][k][n]
}

// ======================= 4. Fused group attention + maxpool ================
// One 4-wave block per (b, group). Wave w owns output columns 16w..16w+15;
// the two row tiles share B fragments and interleave accumulator chains.
__global__ __launch_bounds__(128) void attn1_kernel(
    const float* __restrict__ xyz, const float* __restrict__ points,
    const int* __restrict__ gi, const float* __restrict__ nxyz,
    const half_t* __restrict__ w16, const float* __restrict__ w1p,
    const float* __restrict__ b1p, float* __restrict__ out1) {
  const int b = blockIdx.x / NPOINT;
  const int tid = threadIdx.x, w = tid >> 5;

  __shared__ __align__(16) float  feat32[32][68];
  __shared__ __align__(16) half_t feat16[32][72];
  __shared__ __align__(16) float  posb[32][3];
  __shared__ __align__(16) half_t q16[32][72], k16[32][72];
  __shared__ __align__(16) half_t v16t[64][40];  // transposed: v16t[d][s]
  __shared__ __align__(16) float  tmp32[32][68];
  __shared__ __align__(16) float  S32[32][33];
  __shared__ __align__(16) half_t P16[32][40];
  __shared__ __align__(16) half_t O16[32][72];
  __shared__ int sgi[32];

  if (tid < 32) sgi[tid] = gi[(size_t)blockIdx.x * NSAMPLE + tid];
  const float c0 = nxyz[(size_t)blockIdx.x * 3 + 0];
  const float c1 = nxyz[(size_t)blockIdx.x * 3 + 1];
  const float c2 = nxyz[(size_t)blockIdx.x * 3 + 2];
  __syncthreads();

  const float* xb = xyz + (size_t)b * 3 * N_;
  const float* pb = points + (size_t)b * DIN * N_;
  for (int e = tid; e < 32 * 64; e += 128) {
    int s = e >> 6, d = e & 63, n = sgi[s];
    float val;
    if (d < 3) {
      float xv = xb[d * N_ + n];
      posb[s][d] = xv;
      val = xv - (d == 0 ? c0 : (d == 1 ? c1 : c2));
    } else {
      val = pb[(d - 3) * N_ + n];
    }
    feat32[s][d] = val;
    feat16[s][d] = (half_t)val;
  }
  __syncthreads();

  // hoisted A fragments of feat (rows 0-15 and 16-31, two k-steps)
  const v16h fA00 = load_A16(&feat16[0][0], 72);
  const v16h fA01 = load_A16(&feat16[0][32], 72);
  const v16h fA10 = load_A16(&feat16[16][0], 72);
  const v16h fA11 = load_A16(&feat16[16][32], 72);

  // q = F@Wq, k = F@Wk, v(pre) = F@Wv   (weights stored transposed)
  for (int m = 0; m < 3; ++m) {
    const half_t* wm = w16 + m * 4096 + 16 * w * 64;
    v16h B0 = load_Bc(wm, 64);
    v16h B1 = load_Bc(wm + 32, 64);
    v8f acc0 = {}, acc1 = {};
    acc0 = wmma16(fA00, B0, acc0);
    acc1 = wmma16(fA10, B0, acc1);
    acc0 = wmma16(fA01, B1, acc0);
    acc1 = wmma16(fA11, B1, acc1);
    if (m == 0) {
      store_C16(&q16[0][16 * w], 72, acc0);
      store_C16(&q16[16][16 * w], 72, acc1);
    } else if (m == 1) {
      store_C16(&k16[0][16 * w], 72, acc0);
      store_C16(&k16[16][16 * w], 72, acc1);
    } else {
      store_C32(&tmp32[0][16 * w], 68, acc0);
      store_C32(&tmp32[16][16 * w], 68, acc1);
    }
  }
  __syncthreads();
  // v += pos @ Wp + bp ; store transposed v16t[d][s]
  for (int e = tid; e < 32 * 64; e += 128) {
    int s = e >> 6, d = e & 63;
    float val = tmp32[s][d] + b1p[d] + posb[s][0] * w1p[d] +
                posb[s][1] * w1p[64 + d] + posb[s][2] * w1p[128 + d];
    v16t[d][s] = (half_t)val;
  }
  __syncthreads();

  // S = q @ k^T (32x32): one 16x16 tile per wave
  {
    int r = w >> 1, c = w & 1;
    v8f acc = {};
    acc = wmma16(load_A16(&q16[16 * r][0], 72), load_Bc(&k16[16 * c][0], 72), acc);
    acc = wmma16(load_A16(&q16[16 * r][32], 72), load_Bc(&k16[16 * c][32], 72), acc);
    store_C32(&S32[16 * r][16 * c], 33, acc);
  }
  __syncthreads();

  // row softmax (stable), scale = 1/sqrt(64)
  if (tid < 32) {
    const float scale = 0.125f;
    float mx = -1e30f;
    for (int j = 0; j < 32; ++j) mx = fmaxf(mx, S32[tid][j] * scale);
    float sum = 0.f;
    for (int j = 0; j < 32; ++j) {
      float p = __expf(S32[tid][j] * scale - mx);
      S32[tid][j] = p;
      sum += p;
    }
    float inv = 1.0f / sum;
    for (int j = 0; j < 32; ++j) P16[tid][j] = (half_t)(S32[tid][j] * inv);
  }
  __syncthreads();

  // O = P @ v   (B fragment contiguous from transposed v)
  {
    v16h Bv = load_Bc(&v16t[16 * w][0], 40);
    v8f o0 = {}, o1 = {};
    o0 = wmma16(load_A16(&P16[0][0], 40), Bv, o0);
    o1 = wmma16(load_A16(&P16[16][0], 40), Bv, o1);
    store_C16(&O16[0][16 * w], 72, o0);
    store_C16(&O16[16][16 * w], 72, o1);
  }
  __syncthreads();

  // Y = feat + O @ Wo  (residual rides in as the C fragment)
  {
    const half_t* wo = w16 + 3 * 4096 + 16 * w * 64;
    v16h B0 = load_Bc(wo, 64);
    v16h B1 = load_Bc(wo + 32, 64);
    v8f y0 = load_C32(&feat32[0][16 * w], 68);
    v8f y1 = load_C32(&feat32[16][16 * w], 68);
    y0 = wmma16(load_A16(&O16[0][0], 72), B0, y0);
    y1 = wmma16(load_A16(&O16[16][0], 72), B0, y1);
    y0 = wmma16(load_A16(&O16[0][32], 72), B1, y0);
    y1 = wmma16(load_A16(&O16[16][32], 72), B1, y1);
    store_C32(&tmp32[0][16 * w], 68, y0);
    store_C32(&tmp32[16][16 * w], 68, y1);
  }
  __syncthreads();

  // max-pool over the 32 samples
  if (tid < 64) {
    float mx = tmp32[0][tid];
    for (int s = 1; s < 32; ++s) mx = fmaxf(mx, tmp32[s][tid]);
    out1[(size_t)blockIdx.x * 64 + tid] = mx;
  }
}

// ======================= 5. q/k/v for global attention =====================
// v stored transposed: v2t[b][d][token]
__global__ __launch_bounds__(128) void qkv2_kernel(
    const float* __restrict__ out1, const float* __restrict__ nxyz,
    const half_t* __restrict__ w16, const float* __restrict__ w2p,
    const float* __restrict__ b2p, half_t* __restrict__ q2,
    half_t* __restrict__ k2, half_t* __restrict__ v2t) {
  const int b = blockIdx.x >> 6;
  const int r0 = (blockIdx.x & 63) * 32;
  const int tid = threadIdx.x, w = tid >> 5;
  __shared__ __align__(16) half_t f16s[32][72];
  __shared__ __align__(16) float  tmp32[32][68];
  __shared__ __align__(16) float  posb[32][3];
  const size_t rowbase = (size_t)b * NPOINT + r0;

  for (int e = tid; e < 32 * 64; e += 128) {
    int s = e >> 6, d = e & 63;
    f16s[s][d] = (half_t)out1[(rowbase + s) * 64 + d];
    if (d < 3) posb[s][d] = nxyz[(rowbase + s) * 3 + d];
  }
  __syncthreads();

  const v16h fA00 = load_A16(&f16s[0][0], 72);
  const v16h fA01 = load_A16(&f16s[0][32], 72);
  const v16h fA10 = load_A16(&f16s[16][0], 72);
  const v16h fA11 = load_A16(&f16s[16][32], 72);

  for (int m = 0; m < 3; ++m) {
    const half_t* wm = w16 + (4 + m) * 4096 + 16 * w * 64;
    v16h B0 = load_Bc(wm, 64);
    v16h B1 = load_Bc(wm + 32, 64);
    v8f acc0 = {}, acc1 = {};
    acc0 = wmma16(fA00, B0, acc0);
    acc1 = wmma16(fA10, B0, acc1);
    acc0 = wmma16(fA01, B1, acc0);
    acc1 = wmma16(fA11, B1, acc1);
    if (m == 0) {
      store_C16(q2 + rowbase * 64 + 16 * w, 64, acc0);
      store_C16(q2 + (rowbase + 16) * 64 + 16 * w, 64, acc1);
    } else if (m == 1) {
      store_C16(k2 + rowbase * 64 + 16 * w, 64, acc0);
      store_C16(k2 + (rowbase + 16) * 64 + 16 * w, 64, acc1);
    } else {
      store_C32(&tmp32[0][16 * w], 68, acc0);
      store_C32(&tmp32[16][16 * w], 68, acc1);
    }
  }
  __syncthreads();
  for (int e = tid; e < 32 * 64; e += 128) {
    int d = e >> 5, s = e & 31;  // s fastest -> coalesced v2t rows
    float val = tmp32[s][d] + b2p[d] + posb[s][0] * w2p[d] +
                posb[s][1] * w2p[64 + d] + posb[s][2] * w2p[128 + d];
    v2t[((size_t)b * 64 + d) * NPOINT + (r0 + s)] = (half_t)val;
  }
}

// ======================= 6. Flash attention over 2048 tokens ===============
// 32 query rows per block; O accumulator lives in registers (2 tiles/wave),
// online-softmax rescale folded into the register fragments.
__global__ __launch_bounds__(128) void attn2_kernel(
    const half_t* __restrict__ q2, const half_t* __restrict__ k2,
    const half_t* __restrict__ v2t, const half_t* __restrict__ w16,
    const float* __restrict__ out1, float* __restrict__ out2) {
  const int b = blockIdx.x >> 6;
  const int r0 = (blockIdx.x & 63) * 32;
  const int tid = threadIdx.x, w = tid >> 5, l = lane_id();
  __shared__ __align__(16) half_t qs[32][72];  // q tile, later normalized O16
  __shared__ __align__(16) float  Y32[32][68];
  __shared__ __align__(16) float  S32[32][33];
  __shared__ __align__(16) half_t P16[32][40];
  __shared__ float mrow[32], lrow[32], arow[32];
  const size_t rowbase = (size_t)b * NPOINT + r0;

  for (int e = tid; e < 32 * 64; e += 128) {
    int s = e >> 6, d = e & 63;
    qs[s][d] = q2[(rowbase + s) * 64 + d];
  }
  if (tid < 32) { mrow[tid] = -1e30f; lrow[tid] = 0.f; }
  __syncthreads();

  const int rS = w >> 1, cS = w & 1;
  const v16h a0 = load_A16(&qs[16 * rS][0], 72);
  const v16h a1 = load_A16(&qs[16 * rS][32], 72);
  const float scale = 0.125f;
  v8f accO0 = {}, accO1 = {};  // register-resident O tiles (rows 0-15 / 16-31)

  for (int kt = 0; kt < NPOINT / 32; ++kt) {
    const half_t* kbp = k2 + ((size_t)b * NPOINT + kt * 32 + 16 * cS) * 64;
    v8f sa = {};
    sa = wmma16(a0, load_Bc(kbp, 64), sa);
    sa = wmma16(a1, load_Bc(kbp + 32, 64), sa);
    store_C32(&S32[16 * rS][16 * cS], 33, sa);
    __syncthreads();

    if (tid < 32) {
      float mold = mrow[tid], mx = mold;
      for (int j = 0; j < 32; ++j) mx = fmaxf(mx, S32[tid][j] * scale);
      float al = __expf(mold - mx);
      float sum = 0.f;
      for (int j = 0; j < 32; ++j) {
        float p = __expf(S32[tid][j] * scale - mx);
        P16[tid][j] = (half_t)p;
        sum += p;
      }
      lrow[tid] = lrow[tid] * al + sum;
      arow[tid] = al;
      mrow[tid] = mx;
    }
    __syncthreads();

#pragma unroll
    for (int j = 0; j < 8; ++j) {  // fold online-softmax rescale into O regs
      int rr = 8 * (l >> 4) + j;
      accO0[j] *= arow[rr];
      accO1[j] *= arow[rr + 16];
    }
    v16h Bv = load_Bc(v2t + ((size_t)b * 64 + 16 * w) * NPOINT + kt * 32, NPOINT);
    accO0 = wmma16(load_A16(&P16[0][0], 40), Bv, accO0);
    accO1 = wmma16(load_A16(&P16[16][0], 40), Bv, accO1);
    __syncthreads();
  }

  // normalize O -> f16 into qs (each wave owns columns 16w..16w+15)
  {
    int colO = 16 * w + (l & 15);
#pragma unroll
    for (int j = 0; j < 8; ++j) {
      int rr = 8 * (l >> 4) + j;
      qs[rr][colO] = (half_t)(accO0[j] / lrow[rr]);
      qs[rr + 16][colO] = (half_t)(accO1[j] / lrow[rr + 16]);
    }
  }
  __syncthreads();

  // Y = feat2 + O @ Wo2 (residual loaded straight from global as C fragment)
  {
    const half_t* wo2 = w16 + 7 * 4096 + 16 * w * 64;
    v16h B0 = load_Bc(wo2, 64);
    v16h B1 = load_Bc(wo2 + 32, 64);
    v8f y0 = load_C32(out1 + rowbase * 64 + 16 * w, 64);
    v8f y1 = load_C32(out1 + (rowbase + 16) * 64 + 16 * w, 64);
    y0 = wmma16(load_A16(&qs[0][0], 72), B0, y0);
    y1 = wmma16(load_A16(&qs[16][0], 72), B0, y1);
    y0 = wmma16(load_A16(&qs[0][32], 72), B1, y0);
    y1 = wmma16(load_A16(&qs[16][32], 72), B1, y1);
    store_C32(&Y32[0][16 * w], 68, y0);
    store_C32(&Y32[16][16 * w], 68, y1);
  }
  __syncthreads();

  // write (B, D, NPOINT) transposed, 32-float runs per d
  for (int e = tid; e < 32 * 64; e += 128) {
    int d = e >> 5, s = e & 31;
    out2[((size_t)b * 64 + d) * NPOINT + (r0 + s)] = Y32[s][d];
  }
}

// ======================= launch =======================
extern "C" void kernel_launch(void* const* d_in, const int* in_sizes, int n_in,
                              void* d_out, int out_size, void* d_ws,
                              size_t ws_size, hipStream_t stream) {
  (void)in_sizes; (void)n_in; (void)out_size; (void)ws_size;
  const float* xyz    = (const float*)d_in[0];
  const float* points = (const float*)d_in[1];
  const float* w1q = (const float*)d_in[2];
  const float* w1k = (const float*)d_in[3];
  const float* w1v = (const float*)d_in[4];
  const float* w1o = (const float*)d_in[5];
  const float* w1p = (const float*)d_in[6];
  const float* b1p = (const float*)d_in[7];
  const float* w2q = (const float*)d_in[8];
  const float* w2k = (const float*)d_in[9];
  const float* w2v = (const float*)d_in[10];
  const float* w2o = (const float*)d_in[11];
  const float* w2p = (const float*)d_in[12];
  const float* b2p = (const float*)d_in[13];

  char* ws = (char*)d_ws;
  size_t off = 0;
  auto alloc = [&](size_t bytes) {
    void* p = ws + off;
    off += (bytes + 255) & ~(size_t)255;
    return p;
  };
  int*    fidx = (int*)   alloc((size_t)B_ * NPOINT * 4);
  float*  nxyz = (float*) alloc((size_t)B_ * NPOINT * 3 * 4);
  int*    gi   = (int*)   alloc((size_t)B_ * NPOINT * NSAMPLE * 4);
  float*  out1 = (float*) alloc((size_t)B_ * NPOINT * DD * 4);
  half_t* w16  = (half_t*)alloc((size_t)8 * 4096 * 2);
  half_t* q2   = (half_t*)alloc((size_t)B_ * NPOINT * DD * 2);
  half_t* k2   = (half_t*)alloc((size_t)B_ * NPOINT * DD * 2);
  half_t* v2t  = (half_t*)alloc((size_t)B_ * NPOINT * DD * 2);

  float* out0 = (float*)d_out;                   // (B, 3, NPOINT)
  float* out2 = out0 + (size_t)B_ * 3 * NPOINT;  // (B, D, NPOINT)

  fps_kernel<<<B_, 1024, 0, stream>>>(xyz, fidx, nxyz, out0);
  ballq_kernel<<<B_ * NPOINT / 8, 256, 0, stream>>>(xyz, nxyz, gi);
  wprep_kernel<<<(8 * 4096 + 255) / 256, 256, 0, stream>>>(
      w1q, w1k, w1v, w1o, w2q, w2k, w2v, w2o, w16);
  attn1_kernel<<<B_ * NPOINT, 128, 0, stream>>>(xyz, points, gi, nxyz, w16, w1p,
                                                b1p, out1);
  qkv2_kernel<<<B_ * 64, 128, 0, stream>>>(out1, nxyz, w16, w2p, b2p, q2, k2,
                                           v2t);
  attn2_kernel<<<B_ * 64, 128, 0, stream>>>(q2, k2, v2t, w16, out1, out2);
}
